// GLSS_88862873354245
// MI455X (gfx1250) — matrix operand order repaired and checked by reading, not compiled, not run on hardware
//
#include <hip/hip_runtime.h>
#include <hip/hip_bf16.h>

// ---------------------------------------------------------------------------
// Problem constants (from reference setup_inputs)
// ---------------------------------------------------------------------------
#define BB   4
#define CC   96
#define HH   64
#define WWd  64
#define LL   4096      // H*W
#define DD   192
#define KK   4
#define NN   16
#define RR   6
#define XDIM 38        // R + 2N
#define XDP  48        // XDIM padded to multiple of 16
#define CH   256       // scan chunk length
#define NCH  16        // chunks per row (LL/CH)

typedef __attribute__((ext_vector_type(8)))  _Float16 v8h;
typedef __attribute__((ext_vector_type(16))) _Float16 v16h;
typedef __attribute__((ext_vector_type(8)))  float    v8f;

#define SHUF16(lo, hi) __builtin_shufflevector(lo, hi, 0,1,2,3,4,5,6,7,8,9,10,11,12,13,14,15)

// ---------------------------------------------------------------------------
// Weight pre-pack: f32 row-major [M,K] -> f16 row-major [Mp,K], zero pad rows.
// ---------------------------------------------------------------------------
__global__ __launch_bounds__(256)
void pack_w_kernel(const float* __restrict__ in, _Float16* __restrict__ outp,
                   int M, int K, int Mp, int nz)
{
    long idx = (long)blockIdx.x * blockDim.x + threadIdx.x;
    const long per = (long)Mp * K;
    if (idx >= per * nz) return;
    const int z = (int)(idx / per);
    const long r = idx % per;
    const int m = (int)(r / K);
    const int k = (int)(r % K);
    float v = 0.f;
    if (m < M) v = in[(long)z * M * K + (long)m * K + k];
    outp[idx] = (_Float16)v;
}

// ---------------------------------------------------------------------------
// Transpose-convert: f32 [Ch, L] per batch -> f16 [L, Ch] per batch
// ---------------------------------------------------------------------------
__global__ __launch_bounds__(256)
void t_cvt_kernel(const float* __restrict__ in, _Float16* __restrict__ out, int Ch, long total)
{
    long idx = (long)blockIdx.x * blockDim.x + threadIdx.x;  // b*Ch*L
    if (idx >= total) return;
    const int l = (int)(idx % LL);
    const long bc = idx / LL;
    const int c = (int)(bc % Ch);
    const int b = (int)(bc / Ch);
    out[((long)b * LL + l) * Ch + c] = (_Float16)in[idx];
}

// ---------------------------------------------------------------------------
// WMMA GEMM:  Y[m,n] = epilogue( sum_k A[m,k] * X[k,n] )
//   A16 : f16 row-major [Mp,K], pre-padded (no guards in K loop)
//   Xt  : f16 transposed activations [Ncols, K] per z  (B fragment = 2 b128)
//   Y   : f32 row-major [M, Ncols] per z
//   Y16t: f16 transposed [Ncols, M] per z (one b128 store per tile)
//   flags: 1=relu6  2=accumulate(Y+=)  4=write f32 Y  8=write f16 Y16t
//          16=per-row scale  32=per-row bias
// Wave -> 16x32 output (two 16x16 tiles sharing the A fragment).
// ---------------------------------------------------------------------------
__global__ __launch_bounds__(128)
void gemm_wmma_kernel(const _Float16* __restrict__ A16, const _Float16* __restrict__ Xt,
                      float* __restrict__ Y, _Float16* __restrict__ Y16t,
                      const float* __restrict__ scale, const float* __restrict__ bias,
                      int M, int Kd, int Ncols,
                      int a_mod, long Astride, long Xstride, long Ystride, long Y16stride,
                      int flags)
{
    const int wave = threadIdx.x >> 5;
    const int lane = threadIdx.x & 31;
    const int hi   = lane >> 4;
    const int lm   = lane & 15;
    const int m0   = blockIdx.x * 16;
    const int n0   = (blockIdx.y * 4 + wave) * 32;
    const int z    = blockIdx.z;

    const _Float16* arow = A16 + (long)(z % a_mod) * Astride + (long)(m0 + lm) * Kd;
    const _Float16* b0r  = Xt + (long)z * Xstride + (long)(n0 + lm) * Kd;
    const _Float16* b1r  = b0r + 16L * Kd;
    const long ybase   = (long)z * Ystride;
    const long y16base = (long)z * Y16stride;

    v8f c0 = {}, c1 = {};
    for (int k0 = 0; k0 < Kd; k0 += 32) {
        const v8h al  = *(const v8h*)(arow + k0 + 16 * hi);
        const v8h ah  = *(const v8h*)(arow + k0 + 16 * hi + 8);
        const v8h b0l = *(const v8h*)(b0r + k0 + 16 * hi);
        const v8h b0h = *(const v8h*)(b0r + k0 + 16 * hi + 8);
        const v8h b1l = *(const v8h*)(b1r + k0 + 16 * hi);
        const v8h b1h = *(const v8h*)(b1r + k0 + 16 * hi + 8);
        const v16h a  = SHUF16(al, ah);
        const v16h b0 = SHUF16(b0l, b0h);
        const v16h b1 = SHUF16(b1l, b1h);
        c0 = __builtin_amdgcn_wmma_f32_16x16x32_f16(false, a, false, b0, (short)0, c0, false, false);
        c1 = __builtin_amdgcn_wmma_f32_16x16x32_f16(false, a, false, b1, (short)0, c1, false, false);
    }

#pragma unroll
    for (int t = 0; t < 2; ++t) {
        const v8f cc = t ? c1 : c0;
        const int nn = n0 + t * 16 + lm;
        v8h ov;
#pragma unroll
        for (int r = 0; r < 8; ++r) {
            const int mr = m0 + 8 * hi + r;
            float v = cc[r];
            if (flags & 16) v *= scale[mr];
            if (flags & 32) v += bias[mr];
            const long off = ybase + (long)mr * Ncols + nn;
            if (flags & 2)  v += Y[off];
            if (flags & 1)  v = fminf(fmaxf(v, 0.f), 6.f);
            if ((flags & 4) && mr < M) Y[off] = v;
            ov[r] = (_Float16)v;
        }
        if (flags & 8)
            *(v8h*)(Y16t + y16base + (long)nn * M + m0 + 8 * hi) = ov;
    }
}

// ---------------------------------------------------------------------------
// Depthwise KSxKS conv + bias (+ optional SiLU)
//   out32: f32 row-major [Cn, L];  out16: f16 transposed [L, Cn]
// ---------------------------------------------------------------------------
template <int KS, bool SILU>
__global__ __launch_bounds__(256)
void dwconv_kernel(const float* __restrict__ in, const float* __restrict__ w,
                   const float* __restrict__ bias,
                   float* __restrict__ out32, _Float16* __restrict__ out16,
                   int Cn, long inBstride, long outBstride)
{
    const int PAD = KS / 2;
    long idx = (long)blockIdx.x * blockDim.x + threadIdx.x;
    const long total = (long)BB * Cn * LL;
    if (idx >= total) return;
    const int x = (int)(idx & 63);
    const int y = (int)((idx >> 6) & 63);
    const int c = (int)((idx >> 12) % Cn);
    const int b = (int)((idx >> 12) / Cn);

    const float* ip = in + (long)b * inBstride + (long)c * LL;
    const float* wp = w + c * KS * KS;
    float s = bias[c];
#pragma unroll
    for (int ky = 0; ky < KS; ++ky) {
        const int yy = y + ky - PAD;
        if (yy < 0 || yy >= HH) continue;
#pragma unroll
        for (int kx = 0; kx < KS; ++kx) {
            const int xx = x + kx - PAD;
            if (xx < 0 || xx >= WWd) continue;
            s += wp[ky * KS + kx] * ip[yy * WWd + xx];
        }
    }
    if (SILU) s = s / (1.f + __expf(-s));
    const int l = y * WWd + x;
    if (out32) out32[(long)b * outBstride + (long)c * LL + l] = s;
    if (out16) out16[(long)b * outBstride + (long)l * Cn + c] = (_Float16)s;
}

// ---------------------------------------------------------------------------
// Build 4 scan directions, transposed layout only: xst [b][k][l][d]
//   (coalesced u-reads for the scan AND B-matrix for the x_proj WMMA GEMM)
//   k0 = row-major, k1 = transposed, k2 = reversed(k0), k3 = reversed(k1)
// ---------------------------------------------------------------------------
__global__ __launch_bounds__(256)
void build_xs_kernel(const float* __restrict__ xc, _Float16* __restrict__ xst)
{
    long idx = (long)blockIdx.x * blockDim.x + threadIdx.x;  // over B*D*L
    if (idx >= (long)BB * DD * LL) return;
    const int l = (int)(idx & (LL - 1));
    const long bd = idx >> 12;
    const int d = (int)(bd % DD);
    const int b = (int)(bd / DD);
    const _Float16 hv = (_Float16)xc[idx];
    const int hrow = l >> 6, wcol = l & 63;
    const int pos[4] = { l, wcol * HH + hrow, LL - 1 - l, LL - 1 - (wcol * HH + hrow) };
#pragma unroll
    for (int k = 0; k < 4; ++k)
        xst[(((long)b * KK + k) * LL + pos[k]) * DD + d] = hv;
}

// ---------------------------------------------------------------------------
// Blocked selective scan, pass 1: scan each chunk from h=0, record the
// per-state chunk endpoint h_end and sum(delta). One lane per (b,k,d) row,
// one wave-chunk per block. 1536 independent waves.
// ---------------------------------------------------------------------------
__global__ __launch_bounds__(32)
void scan_pass1_kernel(const _Float16* __restrict__ xst, const float* __restrict__ xdbl,
                       const float* __restrict__ dtw, const float* __restrict__ dtb,
                       const float* __restrict__ Alog,
                       float* __restrict__ chkH, float* __restrict__ chkS)
{
    const int lane = threadIdx.x;
    const int ch = blockIdx.x % NCH;
    const int t  = blockIdx.x / NCH;   // bk*6 + j
    const int j  = t % 6;
    const int bk = t / 6;
    const int k  = bk % KK;
    const int d  = j * 32 + lane;

    float A[NN], h[NN];
#pragma unroll
    for (int n = 0; n < NN; ++n) {
        A[n] = -__expf(Alog[((long)k * DD + d) * NN + n]);
        h[n] = 0.f;
    }
    float wr[RR];
#pragma unroll
    for (int r = 0; r < RR; ++r) wr[r] = dtw[((long)k * DD + d) * RR + r];
    const float bias = dtb[k * DD + d];

    const _Float16* u  = xst  + (long)bk * LL * DD;   // [l][d]
    const float*    xb = xdbl + (long)bk * XDIM * LL;
    const int l0 = ch * CH;

    float sd = 0.f;
    for (int l = l0; l < l0 + CH; ++l) {
        __builtin_prefetch(&u[(long)(l + 32) * DD + d], 0, 1);
        const float uv = (float)u[(long)l * DD + d];
        float s = bias;
#pragma unroll
        for (int r = 0; r < RR; ++r) s += wr[r] * xb[r * LL + l];
        const float delta = (s > 20.f) ? s : log1pf(__expf(s));
        sd += delta;
        const float du = delta * uv;
#pragma unroll
        for (int n = 0; n < NN; ++n)
            h[n] = __expf(delta * A[n]) * h[n] + du * xb[(RR + n) * LL + l];
    }
    const long rbase = ((long)bk * DD + d) * NCH + ch;
#pragma unroll
    for (int n = 0; n < NN; ++n) chkH[rbase * NN + n] = h[n];
    chkS[rbase] = sd;
}

// ---------------------------------------------------------------------------
// Chain: compose chunk carries serially (16 steps), replacing each chunk's
// h_end record with its true START state:  s_{i+1} = exp(A*S_i)*s_i + hend_i
// ---------------------------------------------------------------------------
__global__ __launch_bounds__(32)
void scan_chain_kernel(const float* __restrict__ Alog,
                       float* __restrict__ chkH, const float* __restrict__ chkS)
{
    const int lane = threadIdx.x;
    const int t  = blockIdx.x;    // bk*6 + j
    const int j  = t % 6;
    const int bk = t / 6;
    const int k  = bk % KK;
    const int d  = j * 32 + lane;

    float A[NN], hc[NN];
#pragma unroll
    for (int n = 0; n < NN; ++n) {
        A[n] = -__expf(Alog[((long)k * DD + d) * NN + n]);
        hc[n] = 0.f;
    }
    const long rbase = ((long)bk * DD + d) * NCH;
    for (int i = 0; i < NCH; ++i) {
        const float S = chkS[rbase + i];
        const long cb = (rbase + i) * NN;
#pragma unroll
        for (int n = 0; n < NN; ++n) {
            const float he = chkH[cb + n];
            const float hs = hc[n];
            chkH[cb + n] = hs;                       // true start state
            hc[n] = __expf(A[n] * S) * hs + he;      // carry to next chunk
        }
    }
}

// ---------------------------------------------------------------------------
// Pass 2: rescan each chunk from its true start state, emitting
// outyT[bk][l][d] = y + Ds*u  (lane-contiguous d -> coalesced stores)
// ---------------------------------------------------------------------------
__global__ __launch_bounds__(32)
void scan_pass2_kernel(const _Float16* __restrict__ xst, const float* __restrict__ xdbl,
                       const float* __restrict__ dtw, const float* __restrict__ dtb,
                       const float* __restrict__ Alog, const float* __restrict__ Dsp,
                       const float* __restrict__ chkH, float* __restrict__ outyT)
{
    const int lane = threadIdx.x;
    const int ch = blockIdx.x % NCH;
    const int t  = blockIdx.x / NCH;
    const int j  = t % 6;
    const int bk = t / 6;
    const int k  = bk % KK;
    const int d  = j * 32 + lane;

    float A[NN], h[NN];
    const long rbase = ((long)bk * DD + d) * NCH + ch;
#pragma unroll
    for (int n = 0; n < NN; ++n) {
        A[n] = -__expf(Alog[((long)k * DD + d) * NN + n]);
        h[n] = chkH[rbase * NN + n];
    }
    float wr[RR];
#pragma unroll
    for (int r = 0; r < RR; ++r) wr[r] = dtw[((long)k * DD + d) * RR + r];
    const float bias = dtb[k * DD + d];
    const float Dsd  = Dsp[k * DD + d];

    const _Float16* u  = xst  + (long)bk * LL * DD;
    const float*    xb = xdbl + (long)bk * XDIM * LL;
    float*          o  = outyT + (long)bk * LL * DD;
    const int l0 = ch * CH;

    for (int l = l0; l < l0 + CH; ++l) {
        __builtin_prefetch(&u[(long)(l + 32) * DD + d], 0, 1);
        __builtin_prefetch(&xb[6 * LL + l + 64], 0, 1);
        const float uv = (float)u[(long)l * DD + d];
        float s = bias;
#pragma unroll
        for (int r = 0; r < RR; ++r) s += wr[r] * xb[r * LL + l];
        const float delta = (s > 20.f) ? s : log1pf(__expf(s));
        const float du = delta * uv;
        float y = 0.f;
#pragma unroll
        for (int n = 0; n < NN; ++n) {
            h[n] = __expf(delta * A[n]) * h[n] + du * xb[(RR + n) * LL + l];
            y += h[n] * xb[(RR + NN + n) * LL + l];
        }
        o[(long)l * DD + d] = y + Dsd * uv;
    }
}

// ---------------------------------------------------------------------------
// Combine 4 directions + LayerNorm over D (wave32 shfl reduction) + SiLU(z)
// gate; reads outyT [bk][l][d] and writes y16t [b][l][d] -- both coalesced.
// One wave per pixel.
// ---------------------------------------------------------------------------
__global__ __launch_bounds__(128)
void combine_ln_kernel(const float* __restrict__ outyT, const float* __restrict__ xz,
                       const float* __restrict__ g, const float* __restrict__ be,
                       _Float16* __restrict__ y16t)
{
    const int wave = threadIdx.x >> 5;
    const int lane = threadIdx.x & 31;
    const int pix  = blockIdx.x * 4 + wave;   // b*L + l
    const int b = pix >> 12;
    const int l = pix & (LL - 1);
    const int hrow = l >> 6, wcol = l & 63;
    const int lt = wcol * HH + hrow;
    const long base = (long)b * KK * LL * DD;

    float v[6];
    float sum = 0.f, sq = 0.f;
#pragma unroll
    for (int t = 0; t < 6; ++t) {
        const int d = lane + t * 32;
        const float x =
            outyT[base + ((long)(0 * LL) + l) * DD + d] +
            outyT[base + ((long)(2 * LL) + (LL - 1 - l)) * DD + d] +
            outyT[base + ((long)(1 * LL) + lt) * DD + d] +
            outyT[base + ((long)(3 * LL) + (LL - 1 - lt)) * DD + d];
        v[t] = x;
        sum += x;
        sq  += x * x;
    }
#pragma unroll
    for (int m = 16; m >= 1; m >>= 1) {
        sum += __shfl_xor(sum, m, 32);
        sq  += __shfl_xor(sq,  m, 32);
    }
    const float mu  = sum * (1.f / DD);
    const float var = sq * (1.f / DD) - mu * mu;
    const float inv = rsqrtf(var + 1e-5f);
#pragma unroll
    for (int t = 0; t < 6; ++t) {
        const int d = lane + t * 32;
        const float yn = (v[t] - mu) * inv * g[d] + be[d];
        const float zv = xz[((long)b * (2 * DD) + DD + d) * LL + l];
        const float sz = zv / (1.f + __expf(-zv));
        y16t[((long)b * LL + l) * DD + d] = (_Float16)(yn * sz);
    }
}

// ---------------------------------------------------------------------------
// Host orchestration
// ---------------------------------------------------------------------------
extern "C" void kernel_launch(void* const* d_in, const int* in_sizes, int n_in,
                              void* d_out, int out_size, void* d_ws, size_t ws_size,
                              hipStream_t stream)
{
    const float* x        = (const float*)d_in[0];
    const float* fc1_w    = (const float*)d_in[1];
    const float* bn1_s    = (const float*)d_in[2];
    const float* bn1_b    = (const float*)d_in[3];
    const float* dw3_w    = (const float*)d_in[4];
    const float* dw3_b    = (const float*)d_in[5];
    const float* pw1_w    = (const float*)d_in[6];
    const float* pw1_b    = (const float*)d_in[7];
    const float* dw5_w    = (const float*)d_in[8];
    const float* dw5_b    = (const float*)d_in[9];
    const float* pw2_w    = (const float*)d_in[10];
    const float* pw2_b    = (const float*)d_in[11];
    const float* fc2_w    = (const float*)d_in[12];
    const float* bn2_s    = (const float*)d_in[13];
    const float* bn2_b    = (const float*)d_in[14];
    const float* in_proj_w= (const float*)d_in[15];
    const float* conv_w   = (const float*)d_in[16];
    const float* conv_b   = (const float*)d_in[17];
    const float* xproj_w  = (const float*)d_in[18];
    const float* dtproj_w = (const float*)d_in[19];
    const float* dtproj_b = (const float*)d_in[20];
    const float* A_logs   = (const float*)d_in[21];
    const float* Ds       = (const float*)d_in[22];
    const float* ln_g     = (const float*)d_in[23];
    const float* ln_b     = (const float*)d_in[24];
    const float* out_proj_w = (const float*)d_in[25];
    float* out = (float*)d_out;

    // ---- workspace layout ----
    char* ws = (char*)d_ws;
    size_t off = 0;
    auto alloc = [&](size_t bytes) -> char* {
        char* p = ws + off;
        off = (off + bytes + 255) & ~(size_t)255;
        return p;
    };
    const long BCL = (long)BB * CC * LL;
    const long BDL = (long)BB * DD * LL;

    // packed f16 weights
    _Float16* fc1p  = (_Float16*)alloc((long)CC * CC * 2);
    _Float16* pw1p  = (_Float16*)alloc((long)CC * CC * 2);
    _Float16* pw2p  = (_Float16*)alloc((long)CC * CC * 2);
    _Float16* fc2p  = (_Float16*)alloc((long)CC * CC * 2);
    _Float16* inpp  = (_Float16*)alloc((long)2 * DD * CC * 2);
    _Float16* outpp = (_Float16*)alloc((long)CC * DD * 2);
    _Float16* xprjp = (_Float16*)alloc((long)KK * XDP * DD * 2);
    // activations
    _Float16* x16t  = (_Float16*)alloc(BCL * 2);     // [b][l][C]
    float*    h32   = (float*)   alloc(BCL * 4);     // [b][C][l]
    _Float16* h16t  = (_Float16*)alloc(BCL * 2);     // [b][l][C]
    _Float16* t1t   = (_Float16*)alloc(BCL * 2);
    _Float16* t2t   = (_Float16*)alloc(BCL * 2);
    float*    acc   = (float*)   alloc(BCL * 4);     // [b][C][l]
    _Float16* acc16t= (_Float16*)alloc(BCL * 2);     // [b][l][C]
    float*    xz    = (float*)   alloc(2 * BDL * 4); // [b][384][l]
    float*    xc    = (float*)   alloc(BDL * 4);     // [b][D][l]
    _Float16* xs16t = (_Float16*)alloc((long)KK * BDL * 2);    // [b][k][l][d]
    float*    xdbl  = (float*)   alloc((long)BB * KK * XDIM * LL * 4);
    float*    outyT = (float*)   alloc((long)KK * BDL * 4);    // [b][k][l][d]
    _Float16* y16t  = (_Float16*)alloc(BDL * 2);     // [b][l][D]
    // scan chunk carries
    float*    chkH  = (float*)   alloc((long)BB * KK * DD * NCH * NN * 4);
    float*    chkS  = (float*)   alloc((long)BB * KK * DD * NCH * 4);
    if (off > ws_size) return;

    const dim3 gblk(128, 1, 1);
    const int NT = 32;  // 32 blocks * 4 waves * 32 cols = 4096 columns

    // 0) pack weights to f16 (padded)
    auto packLaunch = [&](const float* src, _Float16* dst, int M, int K, int Mp, int nz) {
        long n = (long)Mp * K * nz;
        pack_w_kernel<<<(int)((n + 255) / 256), 256, 0, stream>>>(src, dst, M, K, Mp, nz);
    };
    packLaunch(fc1_w,      fc1p,  CC,     CC, CC,     1);
    packLaunch(pw1_w,      pw1p,  CC,     CC, CC,     1);
    packLaunch(pw2_w,      pw2p,  CC,     CC, CC,     1);
    packLaunch(fc2_w,      fc2p,  CC,     CC, CC,     1);
    packLaunch(in_proj_w,  inpp,  2 * DD, CC, 2 * DD, 1);
    packLaunch(out_proj_w, outpp, CC,     DD, CC,     1);
    packLaunch(xproj_w,    xprjp, XDIM,   DD, XDP,    KK);

    // 1) x -> f16 transposed
    t_cvt_kernel<<<(int)((BCL + 255) / 256), 256, 0, stream>>>(x, x16t, CC, BCL);

    // 2) fc1 + BN + relu6 -> h32 (f32 rm) + h16t (f16 tr)
    gemm_wmma_kernel<<<dim3(6, NT, BB), gblk, 0, stream>>>(
        fc1p, x16t, h32, h16t, bn1_s, bn1_b, CC, CC, LL,
        1, 0, (long)CC * LL, (long)CC * LL, (long)CC * LL, 1 | 4 | 8 | 16 | 32);

    // 3) depthwise 3x3 / 5x5 (+bias) -> f16 transposed
    dwconv_kernel<3, false><<<(int)((BCL + 255) / 256), 256, 0, stream>>>(
        h32, dw3_w, dw3_b, nullptr, t1t, CC, (long)CC * LL, (long)CC * LL);
    dwconv_kernel<5, false><<<(int)((BCL + 255) / 256), 256, 0, stream>>>(
        h32, dw5_w, dw5_b, nullptr, t2t, CC, (long)CC * LL, (long)CC * LL);

    // 4) pw1 -> acc ; pw2 accumulates
    gemm_wmma_kernel<<<dim3(6, NT, BB), gblk, 0, stream>>>(
        pw1p, t1t, acc, nullptr, nullptr, pw1_b, CC, CC, LL,
        1, 0, (long)CC * LL, (long)CC * LL, 0, 4 | 32);
    gemm_wmma_kernel<<<dim3(6, NT, BB), gblk, 0, stream>>>(
        pw2p, t2t, acc, nullptr, nullptr, pw2_b, CC, CC, LL,
        1, 0, (long)CC * LL, (long)CC * LL, 0, 4 | 32 | 2);

    // 5) in_proj (384x96) -> xz f32 row-major
    gemm_wmma_kernel<<<dim3(24, NT, BB), gblk, 0, stream>>>(
        inpp, h16t, xz, nullptr, nullptr, nullptr, 2 * DD, CC, LL,
        1, 0, (long)CC * LL, (long)2 * DD * LL, 0, 4);

    // 6) depthwise 3x3 on xin + SiLU -> xc
    dwconv_kernel<3, true><<<(int)((BDL + 255) / 256), 256, 0, stream>>>(
        xz, conv_w, conv_b, xc, nullptr, DD, (long)2 * DD * LL, (long)DD * LL);

    // 7) 4-direction xs, transposed layout (GEMM-B + coalesced scan reads)
    build_xs_kernel<<<(int)((BDL + 255) / 256), 256, 0, stream>>>(xc, xs16t);

    // 8) x_proj per direction: (38x192)@(192x4096), M padded to 48
    gemm_wmma_kernel<<<dim3(3, NT, BB * KK), gblk, 0, stream>>>(
        xprjp, xs16t, xdbl, nullptr, nullptr, nullptr, XDIM, DD, LL,
        KK, (long)XDP * DD, (long)DD * LL, (long)XDIM * LL, 0, 4);

    // 9) blocked selective scan: pass1 (1536 waves) -> chain (96) -> pass2 (1536)
    scan_pass1_kernel<<<BB * KK * 6 * NCH, 32, 0, stream>>>(
        xs16t, xdbl, dtproj_w, dtproj_b, A_logs, chkH, chkS);
    scan_chain_kernel<<<BB * KK * 6, 32, 0, stream>>>(A_logs, chkH, chkS);
    scan_pass2_kernel<<<BB * KK * 6 * NCH, 32, 0, stream>>>(
        xs16t, xdbl, dtproj_w, dtproj_b, A_logs, Ds, chkH, outyT);

    // 10) combine directions + LN + SiLU(z) gate -> y16t
    combine_ln_kernel<<<BB * LL / 4, 128, 0, stream>>>(outyT, xz, ln_g, ln_b, y16t);

    // 11) out_proj accumulates into acc, emit f16 transposed (x1+x2+x3)
    gemm_wmma_kernel<<<dim3(6, NT, BB), gblk, 0, stream>>>(
        outpp, y16t, acc, acc16t, nullptr, nullptr, CC, DD, LL,
        1, 0, (long)DD * LL, (long)CC * LL, (long)CC * LL, 2 | 8);

    // 12) fc2 + BN + relu6 -> d_out (f32 row-major [B,C,H,W])
    gemm_wmma_kernel<<<dim3(6, NT, BB), gblk, 0, stream>>>(
        fc2p, acc16t, out, nullptr, bn2_s, bn2_b, CC, CC, LL,
        1, 0, (long)CC * LL, (long)CC * LL, 0, 1 | 4 | 16 | 32);
}
